// FocalStream_8237747274149
// MI455X (gfx1250) — matrix operand order repaired and checked by reading, not compiled, not run on hardware
//
#include <hip/hip_runtime.h>

// ---------------- problem constants ----------------
#define Bn    16
#define Nn    1369
#define Dn    1024
#define NPAD  1376           // 86 * 16
#define QT    86             // 16-row tiles per batch
#define KT32  43             // NPAD / 32
#define HIDN  512
#define INV_SCALE 0.08838834764831845f   // 1/sqrt(128)

// ---------------- WMMA / TDM types ----------------
typedef __attribute__((ext_vector_type(16))) __bf16 v16bf;
typedef __attribute__((ext_vector_type(8)))  __bf16 v8bf;
typedef __attribute__((ext_vector_type(8)))  float  v8f;
typedef __attribute__((ext_vector_type(4)))  unsigned int u32x4;
typedef __attribute__((ext_vector_type(8)))  int i32x8;
typedef __attribute__((ext_vector_type(4)))  int i32x4;

__device__ __forceinline__ unsigned short f2bf(float f) {
    union { float f; unsigned u; } x; x.f = f;
    unsigned u = x.u;
    unsigned r = (u + 0x7FFFu + ((u >> 16) & 1u)) >> 16;
    return (unsigned short)r;
}
__device__ __forceinline__ float bf2f(unsigned short b) {
    union { unsigned u; float f; } x; x.u = ((unsigned)b) << 16;
    return x.f;
}

// Build a 16-element bf16 fragment out of two 16-byte chunks.
__device__ __forceinline__ v16bf mkfrag(const unsigned short* p0, const unsigned short* p1) {
    union { v16bf v; v8bf h[2]; } u;
    u.h[0] = *(const v8bf*)p0;
    u.h[1] = *(const v8bf*)p1;
    return u.v;
}

__device__ __forceinline__ v8f wmma_bf16(v16bf a, v16bf b, v8f c) {
    return __builtin_amdgcn_wmma_f32_16x16x32_bf16(false, a, false, b, (short)0, c, false, false);
}

// TDM: 1-D async copy of `bytes` (<= 65535) from global to LDS. Wave-level op.
// D# group0: count=1, lds_addr, 57-bit global addr, type=2 ("image").
// D# group1: data_size=1B, tensor_dim0 = tile_dim0 = bytes, single row.
// This toolchain's builtin takes 6 args: (g0, g1, g2, g3, extra, cpol).
__device__ __forceinline__ void tdm_copy_1d(const void* gsrc, unsigned lds_off, int bytes) {
    unsigned long long ga = (unsigned long long)(size_t)gsrc;
    u32x4 g0;
    g0[0] = 1u;                                   // count = 1 valid descriptor
    g0[1] = lds_off;                              // LDS byte address
    g0[2] = (unsigned)ga;                         // global addr [31:0]
    g0[3] = (unsigned)((ga >> 32) & 0x1FFFFFFu) | 0x80000000u;   // [56:32] | type=2
    i32x8 g1;
    g1[0] = 0;                                    // data_size=1B, no flags
    g1[1] = (bytes & 0xFFFF) << 16;               // tensor_dim0[15:0]
    g1[2] = ((bytes >> 16) & 0xFFFF) | (1 << 16); // tensor_dim0[31:16] | tensor_dim1=1
    g1[3] = (bytes & 0xFFFF) << 16;               // tile_dim0
    g1[4] = 0;                                    // tile_dim1/2 unused
    g1[5] = bytes;                                // tensor_dim0_stride[31:0]
    g1[6] = 0;
    g1[7] = 0;
    i32x4 g2, g3;
    for (int e = 0; e < 4; ++e) { g2[e] = 0; g3[e] = 0; }
    i32x8 g4;
    for (int e = 0; e < 8; ++e) g4[e] = 0;
    __builtin_amdgcn_tensor_load_to_lds(g0, g1, g2, g3, g4, 0);
}

// 16x16 score tile: S = q_tile(16xD) . k_tile(16xD)^T, software-pipelined over D.
__device__ __forceinline__ v8f score_tile(const unsigned short* qb, const unsigned short* kb,
                                          int batch, int q0, int kt2, int lane) {
    v8f s;
    for (int e = 0; e < 8; ++e) s[e] = 0.f;
    const unsigned short* arow = qb + ((size_t)(batch * NPAD + q0 + (lane & 15))) * Dn + ((lane >> 4) * 8);
    const unsigned short* brow = kb + ((size_t)(batch * NPAD + kt2 * 16 + (lane & 15))) * Dn + ((lane >> 4) * 8);
    v16bf a = mkfrag(arow, arow + 16);
    v16bf b = mkfrag(brow, brow + 16);
    for (int kf = 0; kf < 32; ++kf) {
        v16bf an = a, bn = b;
        if (kf < 31) {                     // prefetch next fragments before the MMA
            an = mkfrag(arow + (kf + 1) * 32, arow + (kf + 1) * 32 + 16);
            bn = mkfrag(brow + (kf + 1) * 32, brow + (kf + 1) * 32 + 16);
        }
        s = wmma_bf16(a, b, s);
        a = an; b = bn;
    }
    return s;
}

// ---------------- K0: zero colsum ----------------
__global__ void k_zero(float* p, int n) {
    int i = blockIdx.x * 256 + threadIdx.x;
    if (i < n) p[i] = 0.f;
}

// ---------------- K1: x = tokens + pos_encoding, bf16, padded ----------------
__global__ void k_prep(const float* __restrict__ tok, unsigned short* __restrict__ xb) {
    int row = blockIdx.x;                 // 0 .. Bn*NPAD-1
    int b = row / NPAD, n = row % NPAD;
    int tid = threadIdx.x;
    unsigned short* xp = xb + (size_t)row * Dn;
    if (n >= Nn) {
        for (int e = 0; e < 4; ++e) xp[tid + 256 * e] = 0;
        return;
    }
    const float* tp = tok + ((size_t)b * Nn + n) * Dn;
    float rowf = (float)(n / 37);
    float colf = (float)(n % 37);
    const float LOGC = 0.03597789207803197f;     // ln(10000)/256
    for (int e = 0; e < 4; ++e) {
        int d = tid + 256 * e;
        float base = (d < 512) ? rowf : colf;
        int dd = (d < 512) ? d : d - 512;
        int j = dd >> 1;
        float div = __expf(-(float)j * LOGC);
        float ang = base * div;
        float pe = (dd & 1) ? __cosf(ang) : __sinf(ang);
        xp[d] = f2bf(tp[d] + pe);
    }
}

// ---------------- K2: pack Wq/Wk/Wv into B-fragment layout, bf16 ----------------
__global__ void k_packw(const float* __restrict__ Wq, const float* __restrict__ Wk,
                        const float* __restrict__ Wv, unsigned short* __restrict__ wpack) {
    int z = blockIdx.y;
    const float* W = (z == 0) ? Wq : (z == 1) ? Wk : Wv;
    unsigned idx = blockIdx.x * 256 + threadIdx.x;
    int r = idx & 15, lane = (idx >> 4) & 31, nt = (idx >> 9) & 63, kt = idx >> 15;
    int k = kt * 32 + (lane >> 4) * 8 + (r & 7) + (r >> 3) * 16;
    int n = nt * 16 + (lane & 15);
    wpack[(size_t)z * Dn * Dn + idx] = f2bf(W[(size_t)k * Dn + n]);
}

// ---------------- K3: q/k/v = x @ W + b ----------------
// B panels (4KB per K-step, shared by all 8 waves) are DMA'd to LDS by the
// Tensor Data Mover, double-buffered, overlapping DMA with the WMMA burst.
__global__ void __launch_bounds__(256)
k_qkv(const unsigned short* __restrict__ xb, const unsigned short* __restrict__ wpack,
      const float* __restrict__ bq, const float* __restrict__ bk, const float* __restrict__ bv,
      unsigned short* __restrict__ qb, unsigned short* __restrict__ kb,
      unsigned short* __restrict__ vT) {
    __shared__ unsigned short bstage[2][2048];    // 2 x 4KB weight panels
    int lane = threadIdx.x & 31, wave = threadIdx.x >> 5;
    int tt = blockIdx.x * 8 + wave;               // 0 .. Bn*QT-1
    int batch = tt / QT, rt = tt % QT;
    int r0 = rt * 16;
    int z = blockIdx.z;
    const unsigned short* wz = wpack + (size_t)z * Dn * Dn;
    const float* bias = (z == 0) ? bq : (z == 1) ? bk : bv;
    int nt0 = blockIdx.y * 4;                     // 4 column tiles of 16

    v8f acc[4];
    for (int t = 0; t < 4; ++t)
        for (int e = 0; e < 8; ++e) acc[t][e] = 0.f;

    const unsigned short* arow =
        xb + ((size_t)(batch * NPAD + r0 + (lane & 15))) * Dn + ((lane >> 4) * 8);

    // prologue: DMA panel kt=0 into buffer 0
    if (wave == 0) {
        tdm_copy_1d(wz + ((size_t)(nt0)) * 512, (unsigned)(size_t)&bstage[0][0], 4096);
        __builtin_amdgcn_s_wait_tensorcnt(0);
    }
    __syncthreads();

    v16bf a_cur = mkfrag(arow, arow + 16);
    for (int kt = 0; kt < 32; ++kt) {
        // kick DMA for the next panel into the other buffer (overlaps with MMAs)
        if (wave == 0 && kt + 1 < 32) {
            tdm_copy_1d(wz + ((size_t)((kt + 1) * 64 + nt0)) * 512,
                        (unsigned)(size_t)&bstage[(kt + 1) & 1][0], 4096);
        }
        // prefetch next A fragment
        v16bf a_nxt = a_cur;
        if (kt + 1 < 32)
            a_nxt = mkfrag(arow + (kt + 1) * 32, arow + (kt + 1) * 32 + 16);

        const unsigned short* bl = &bstage[kt & 1][lane * 16];
        for (int t = 0; t < 4; ++t) {
            v16bf bb = mkfrag(bl + t * 512, bl + t * 512 + 8);
            acc[t] = wmma_bf16(a_cur, bb, acc[t]);
        }
        a_cur = a_nxt;

        if (wave == 0 && kt + 1 < 32) __builtin_amdgcn_s_wait_tensorcnt(0);
        __syncthreads();
    }

    int mrow = (lane >> 4) * 8;
    int coll = lane & 15;
    for (int t = 0; t < 4; ++t) {
        int col = (nt0 + t) * 16 + coll;
        float bb = bias[col];
        if (z < 2) {
            unsigned short* outp = ((z == 0) ? qb : kb) +
                                   ((size_t)(batch * NPAD + r0 + mrow)) * Dn + col;
            for (int g = 0; g < 8; ++g) outp[(size_t)g * Dn] = f2bf(acc[t][g] + bb);
        } else {
            // vT[batch][feature=col][token] : 8 consecutive tokens, contiguous
            unsigned short* outp = vT + ((size_t)(batch * Dn + col)) * NPAD + r0 + mrow;
            for (int g = 0; g < 8; ++g) outp[g] = f2bf(acc[t][g] + bb);
        }
    }
}

// ---------------- K4: flash pass 1 — per-query running max / sumexp ----------------
__global__ void __launch_bounds__(256)
k_stats(const unsigned short* __restrict__ qb, const unsigned short* __restrict__ kb,
        float* __restrict__ rowmax, float* __restrict__ rowsum) {
    int lane = threadIdx.x & 31, wave = threadIdx.x >> 5;
    int tt = blockIdx.x * 8 + wave;
    int batch = tt / QT, qt = tt % QT, q0 = qt * 16;
    float mr[8], lr[8];
    for (int g = 0; g < 8; ++g) { mr[g] = -3.0e38f; lr[g] = 0.f; }
    for (int kt2 = 0; kt2 < QT; ++kt2) {
        v8f s = score_tile(qb, kb, batch, q0, kt2, lane);
        int key = kt2 * 16 + (lane & 15);
        bool kvalid = key < Nn;
        for (int g = 0; g < 8; ++g) {
            float v = kvalid ? s[g] * INV_SCALE : -3.0e38f;
            float tm = v;
            tm = fmaxf(tm, __shfl_xor(tm, 1, 32));
            tm = fmaxf(tm, __shfl_xor(tm, 2, 32));
            tm = fmaxf(tm, __shfl_xor(tm, 4, 32));
            tm = fmaxf(tm, __shfl_xor(tm, 8, 32));
            float mnew = fmaxf(mr[g], tm);
            float pe = kvalid ? __expf(v - mnew) : 0.f;
            float ps = pe;
            ps += __shfl_xor(ps, 1, 32);
            ps += __shfl_xor(ps, 2, 32);
            ps += __shfl_xor(ps, 4, 32);
            ps += __shfl_xor(ps, 8, 32);
            lr[g] = lr[g] * __expf(mr[g] - mnew) + ps;
            mr[g] = mnew;
        }
    }
    int base = batch * NPAD + q0 + ((lane >= 16) ? 8 : 0);
    if ((lane & 15) == 0) {
        for (int g = 0; g < 8; ++g) { rowmax[base + g] = mr[g]; rowsum[base + g] = lr[g]; }
    }
}

// ---------------- K5: flash pass 2 — p, colsum, diag, attended = p @ v ----------------
__global__ void __launch_bounds__(256)
k_pv(const unsigned short* __restrict__ qb, const unsigned short* __restrict__ kb,
     const unsigned short* __restrict__ vT, const float* __restrict__ rowmax,
     const float* __restrict__ rowsum, float* __restrict__ colsum,
     float* __restrict__ diag, unsigned short* __restrict__ attb) {
    __shared__ unsigned short plds[16][NPAD + 8];   // full probability row-block, bf16
    int lane = threadIdx.x & 31, wave = threadIdx.x >> 5;
    int batch = blockIdx.x / QT, qt = blockIdx.x % QT, q0 = qt * 16;
    int mrow = (lane >> 4) * 8, coll = lane & 15;

    float rm[8], rsv[8];
    int rbase = batch * NPAD + q0 + mrow;
    for (int g = 0; g < 8; ++g) {
        rm[g] = rowmax[rbase + g];
        float l = rowsum[rbase + g];
        rsv[g] = (l > 0.f) ? 1.f / l : 0.f;
    }

    // Phase 1: probabilities -> LDS; colsum (atomics); attention diagonal.
    for (int kt2 = wave; kt2 < QT; kt2 += 8) {
        v8f s = score_tile(qb, kb, batch, q0, kt2, lane);
        int key = kt2 * 16 + coll;
        bool kvalid = key < Nn;
        float p[8];
        float cp = 0.f;
        for (int g = 0; g < 8; ++g) {
            float v = kvalid ? __expf(s[g] * INV_SCALE - rm[g]) * rsv[g] : 0.f;
            p[g] = v;
            if (q0 + mrow + g < Nn) cp += v;            // exclude padded queries
            plds[mrow + g][kt2 * 16 + coll] = f2bf(v);
        }
        cp += __shfl_xor(cp, 16, 32);
        if (lane < 16 && kvalid) atomicAdd(&colsum[batch * NPAD + key], cp);
        if (kt2 == qt) {
            int c = coll;
            if (((lane < 16) == (c < 8)) && (q0 + c) < Nn)
                diag[batch * NPAD + q0 + c] = p[c - mrow];
        }
    }
    __syncthreads();

    // Phase 2: attended[q0..q0+15][:] = p @ v  (A from LDS, B from vT), pipelined
    for (int f = wave; f < 64; f += 8) {
        v8f acc;
        for (int e = 0; e < 8; ++e) acc[e] = 0.f;
        int feat = f * 16 + coll;
        const unsigned short* brow =
            vT + ((size_t)(batch * Dn + feat)) * NPAD + ((lane >> 4) * 8);
        const unsigned short* a00 = &plds[coll][mrow];
        v16bf a = mkfrag(a00, a00 + 16);
        v16bf b = mkfrag(brow, brow + 16);
        for (int ks = 0; ks < KT32; ++ks) {
            v16bf an = a, bn = b;
            if (ks < KT32 - 1) {
                const unsigned short* a0 = a00 + (ks + 1) * 32;
                an = mkfrag(a0, a0 + 16);
                const unsigned short* bp = brow + (ks + 1) * 32;
                bn = mkfrag(bp, bp + 16);
            }
            acc = wmma_bf16(a, b, acc);
            a = an; b = bn;
        }
        unsigned short* outp = attb + ((size_t)(batch * NPAD + q0 + mrow)) * Dn + feat;
        for (int g = 0; g < 8; ++g) outp[(size_t)g * Dn] = f2bf(acc[g]);
    }
}

// ---------------- K6: pa / fallbacks / curiosity modulation / fa ----------------
__device__ __forceinline__ float cbias(int i) {
    float y = (float)(i / 37) - 18.f;
    float x = (float)(i % 37) - 18.f;
    return 0.3f * __expf(-(x * x + y * y) * (18.0f / 1369.0f));
}

__device__ __forceinline__ float blk_reduce(float v, float* sh) {
    int tid = threadIdx.x;
    sh[tid] = v; __syncthreads();
    for (int s = 128; s > 0; s >>= 1) {
        if (tid < s) sh[tid] += sh[tid + s];
        __syncthreads();
    }
    float r = sh[0]; __syncthreads();
    return r;
}

__global__ void __launch_bounds__(256)
k_pa_fa(const float* __restrict__ colsum, const float* __restrict__ diag,
        const float* __restrict__ rowsum, const float* __restrict__ cur,
        const float* __restrict__ Wc1, const float* __restrict__ bc1,
        const float* __restrict__ Wc2, const float* __restrict__ bc2,
        const float* __restrict__ awp, float* __restrict__ pa, float* __restrict__ fa_out) {
    __shared__ float sh[256];
    int b = blockIdx.x, tid = threadIdx.x;

    float c = cur[b];
    float hj[64];
    for (int j = 0; j < 64; ++j) {
        float h = c * Wc1[j] + bc1[j];
        hj[j] = (h > 0.f) ? h : 0.f;
    }
    float cw = 0.f;
    for (int i = 0; i < 8; ++i) {
        float a = bc2[i];
        for (int j = 0; j < 64; ++j) a += hj[j] * Wc2[j * 8 + i];
        cw += 1.f / (1.f + __expf(-a));
    }
    cw *= 0.125f;
    float aw = awp[0];

    const float* cs = colsum + b * NPAD;
    const float* dg = diag + b * NPAD;
    const float* rs = rowsum + b * NPAD;
    float* paw = pa + b * NPAD;

    float s1 = 0.f, s2 = 0.f;
    for (int i = tid; i < Nn; i += 256) {
        float v = cs[i] * (1.0f / (float)Nn) + cbias(i);
        paw[i] = v; s1 += v; s2 += v * v;
    }
    float S1 = blk_reduce(s1, sh);
    float S2 = blk_reduce(s2, sh);
    float var = (S2 - S1 * S1 / (float)Nn) / (float)(Nn - 1);
    if (var < 1e-6f) {                                   // fallback: attention diagonal
        s1 = 0.f; s2 = 0.f;
        for (int i = tid; i < Nn; i += 256) {
            float v = dg[i] + cbias(i);
            paw[i] = v; s1 += v; s2 += v * v;
        }
        S1 = blk_reduce(s1, sh);
        S2 = blk_reduce(s2, sh);
        var = (S2 - S1 * S1 / (float)Nn) / (float)(Nn - 1);
        if (var < 1e-6f) {                               // fallback: attn row max = 1/rowsum
            s1 = 0.f;
            for (int i = tid; i < Nn; i += 256) {
                float l = rs[i];
                float v = ((l > 0.f) ? 1.f / l : 0.f) + cbias(i);
                paw[i] = v; s1 += v;
            }
            S1 = blk_reduce(s1, sh);
        }
    }
    float inv = 1.f / (S1 + 1e-8f);
    float s3 = 0.f;
    for (int i = tid; i < Nn; i += 256) {
        float p = paw[i] * inv;
        float f = p * (1.f + aw * cw);                   // aw*p*(1+cw) + (1-aw)*p
        f = (f < 1e-8f) ? 1e-8f : f;
        paw[i] = f; s3 += f;
    }
    float S3 = blk_reduce(s3, sh);
    float inv3 = 1.f / (S3 + 1e-8f);
    for (int i = tid; i < Nn; i += 256) fa_out[b * Nn + i] = paw[i] * inv3;
}

// ---------------- K7: weighted[b][d] = sum_n attended[b][n][d] * fa[b][n] ----------------
__global__ void __launch_bounds__(256)
k_weighted(const unsigned short* __restrict__ attb, const float* __restrict__ fa,
           float* __restrict__ weighted) {
    int b = blockIdx.x >> 2;
    int d = ((blockIdx.x & 3) << 8) + threadIdx.x;
    const unsigned short* ap = attb + ((size_t)b * NPAD) * Dn + d;
    const float* fap = fa + b * Nn;
    float s = 0.f;
    for (int n = 0; n < Nn; ++n) s += bf2f(ap[(size_t)n * Dn]) * fap[n];
    weighted[b * Dn + d] = s;
}

// ---------------- K8: final MLP ----------------
__global__ void __launch_bounds__(256)
k_mlp1(const float* __restrict__ weighted, const float* __restrict__ Wp1,
       const float* __restrict__ bp1, float* __restrict__ hbuf) {
    int b = blockIdx.x, tid = threadIdx.x;
    const float* w = weighted + b * Dn;
    for (int j = tid; j < HIDN; j += 256) {
        float a = bp1[j];
        for (int d = 0; d < Dn; ++d) a += w[d] * Wp1[(size_t)d * HIDN + j];
        hbuf[b * HIDN + j] = (a > 0.f) ? a : 0.f;
    }
}

__global__ void __launch_bounds__(128)
k_mlp2(const float* __restrict__ hbuf, const float* __restrict__ Wp2,
       const float* __restrict__ bp2, float* __restrict__ outp) {
    int b = blockIdx.x, i = threadIdx.x;
    const float* h = hbuf + b * HIDN;
    float a = bp2[i];
    for (int j = 0; j < HIDN; ++j) a += h[j] * Wp2[j * 128 + i];
    outp[b * 128 + i] = a;
}

// ---------------- host launcher ----------------
extern "C" void kernel_launch(void* const* d_in, const int* in_sizes, int n_in,
                              void* d_out, int out_size, void* d_ws, size_t ws_size,
                              hipStream_t stream) {
    (void)in_sizes; (void)n_in; (void)out_size; (void)ws_size;
    const float* tok = (const float*)d_in[0];
    const float* cur = (const float*)d_in[1];
    const float* Wq  = (const float*)d_in[2];
    const float* bq  = (const float*)d_in[3];
    const float* Wk  = (const float*)d_in[4];
    const float* bk  = (const float*)d_in[5];
    const float* Wv  = (const float*)d_in[6];
    const float* bv  = (const float*)d_in[7];
    const float* Wc1 = (const float*)d_in[8];
    const float* bc1 = (const float*)d_in[9];
    const float* Wc2 = (const float*)d_in[10];
    const float* bc2 = (const float*)d_in[11];
    const float* Wp1 = (const float*)d_in[12];
    const float* bp1 = (const float*)d_in[13];
    const float* Wp2 = (const float*)d_in[14];
    const float* bp2 = (const float*)d_in[15];
    const float* awp = (const float*)d_in[16];
    float* outp = (float*)d_out;
    float* fa_out = outp + Bn * 128;

    char* w = (char*)d_ws;
    size_t off = 0;
    auto take = [&](size_t bytes) -> char* {
        char* p = w + off;
        off += (bytes + 255) & ~(size_t)255;
        return p;
    };
    unsigned short* xb    = (unsigned short*)take((size_t)Bn * NPAD * Dn * 2);
    unsigned short* qb    = (unsigned short*)take((size_t)Bn * NPAD * Dn * 2);
    unsigned short* kb    = (unsigned short*)take((size_t)Bn * NPAD * Dn * 2);
    unsigned short* vT    = (unsigned short*)take((size_t)Bn * Dn * NPAD * 2);
    unsigned short* attb  = (unsigned short*)take((size_t)Bn * NPAD * Dn * 2);
    unsigned short* wpack = (unsigned short*)take((size_t)3 * Dn * Dn * 2);
    float* rowmax   = (float*)take((size_t)Bn * NPAD * 4);
    float* rowsum   = (float*)take((size_t)Bn * NPAD * 4);
    float* colsum   = (float*)take((size_t)Bn * NPAD * 4);
    float* diag     = (float*)take((size_t)Bn * NPAD * 4);
    float* pa       = (float*)take((size_t)Bn * NPAD * 4);
    float* weighted = (float*)take((size_t)Bn * Dn * 4);
    float* hbuf     = (float*)take((size_t)Bn * HIDN * 4);

    k_zero<<<(Bn * NPAD + 255) / 256, 256, 0, stream>>>(colsum, Bn * NPAD);
    k_prep<<<Bn * NPAD, 256, 0, stream>>>(tok, xb);
    k_packw<<<dim3(4096, 3), 256, 0, stream>>>(Wq, Wk, Wv, wpack);
    k_qkv<<<dim3(Bn * QT / 8, Dn / 64, 3), 256, 0, stream>>>(xb, wpack, bq, bk, bv, qb, kb, vT);
    k_stats<<<Bn * QT / 8, 256, 0, stream>>>(qb, kb, rowmax, rowsum);
    k_pv<<<Bn * QT, 256, 0, stream>>>(qb, kb, vT, rowmax, rowsum, colsum, diag, attb);
    k_pa_fa<<<Bn, 256, 0, stream>>>(colsum, diag, rowsum, cur, Wc1, bc1, Wc2, bc2, awp, pa, fa_out);
    k_weighted<<<Bn * 4, 256, 0, stream>>>(attb, fa_out, weighted);
    k_mlp1<<<Bn, 256, 0, stream>>>(weighted, Wp1, bp1, hbuf);
    k_mlp2<<<Bn, 128, 0, stream>>>(hbuf, Wp2, bp2, outp);
}
